// MambaDirectionEstimator_49117245997239
// MI455X (gfx1250) — matrix operand order repaired and checked by reading, not compiled, not run on hardware
//
#include <hip/hip_runtime.h>
#include <hip/hip_bf16.h>
#include <math.h>

// Problem constants (reference: B=32, S=2048, V=32000, D=1024, N=16)
#define Bb 32
#define Ss 2048
#define Dm 1024
#define Nn 16
#define BS (Bb * Ss)         // 65536 rows of M
#define MT 64                // M rows per workgroup (4 WMMA row-tiles)
#define LDX 1032             // padded LDS row stride (elements)

typedef __attribute__((ext_vector_type(16))) __bf16 v16bf;
typedef __attribute__((ext_vector_type(4)))  __bf16 v4bf;
typedef __attribute__((ext_vector_type(8)))  float  v8f;

#if __has_builtin(__builtin_amdgcn_tanhf)
#define FAST_TANH(x) __builtin_amdgcn_tanhf(x)
#else
#define FAST_TANH(x) tanhf(x)
#endif

// ---------------------------------------------------------------------------
// A-fragment from LDS (16x32 MxK bf16, ISA 7.12.2 layout):
// lanes 0-15: M=0..15, K[0..7]+K[16..23]; lanes 16-31: K[8..15]+K[24..31].
__device__ __forceinline__ v16bf load_A(const __bf16* X, int k0, int lane) {
  const int m    = lane & 15;
  const int half = lane >> 4;
  const __bf16* p = X + m * LDX + k0 + half * 8;
  v16bf a;
#pragma unroll
  for (int i = 0; i < 8; ++i) { a[i] = p[i]; a[8 + i] = p[16 + i]; }
  return a;
}

// B-fragment from pre-swizzled bf16 weights: fragment (nt,kt) is stored as
// 32 lanes x 16 bf16, lane-major -> each lane loads 32 contiguous bytes,
// lanes consecutive => fully coalesced, no conversion in the hot loop.
__device__ __forceinline__ v16bf load_Bp(const __bf16* Wp, int nt, int kt, int lane) {
  return *(const v16bf*)(Wp + (((size_t)(nt * (Dm / 32) + kt)) * 32 + lane) * 16);
}

// ---------------------------------------------------------------------------
// One-time weight pack: f32 row-major W[nTiles*16][1024] -> bf16 fragments in
// WMMA B-operand lane order. B(k,n) = W[nt*16+n][kt*32 + half*8 + (i&7) + (i>>3)*16].
__global__ void __launch_bounds__(256)
pack_b_frags(const float* __restrict__ W, __bf16* __restrict__ out) {
  const int g      = blockIdx.x * 256 + threadIdx.x;
  const int frag   = g >> 9;          // 512 elems per fragment
  const int within = g & 511;
  const int lane   = within >> 4;
  const int i      = within & 15;
  const int nt     = frag >> 5;       // frag / 32
  const int kt     = frag & 31;
  const int n      = lane & 15;
  const int half   = lane >> 4;
  const int row    = nt * 16 + n;
  const int col    = kt * 32 + half * 8 + (i & 7) + (i >> 3) * 16;
  out[g] = (__bf16)W[(size_t)row * Dm + col];
}

// ---------------------------------------------------------------------------
// Fused: embedding gather -> gate GEMM (sigmoid(X Wg^T + bg) .* X) -> B-proj.
// One workgroup = 64 rows of M (4 row-tiles); both directions from the same
// X tile. Each B-fragment load feeds 4 WMMAs (64 flops per L2 byte).
__global__ void __launch_bounds__(256)
fused_gate_proj(const int* __restrict__ ids, const float* __restrict__ emb,
                const __bf16* __restrict__ Wgp_f, const float* __restrict__ bg_f,
                const __bf16* __restrict__ WBp_f,
                const __bf16* __restrict__ Wgp_b, const float* __restrict__ bg_b,
                const __bf16* __restrict__ WBp_b,
                float* __restrict__ Bt_f, float* __restrict__ Bt_b) {
  __shared__ __bf16 Xlds[MT * LDX];   // 132 KB
  __shared__ __bf16 Ylds[MT * LDX];   // 132 KB
  __shared__ float  red[8 * 256];     // 8 KB

  const int t    = threadIdx.x;
  const int lane = t & 31;
  const int wave = t >> 5;
  const size_t m0 = (size_t)blockIdx.x * MT;

  // Gather 64 embedding rows (f32 -> bf16) into LDS.
  for (int i = 0; i < MT; ++i) {
    const int id = ids[m0 + i];
    float4 v = ((const float4*)(emb + (size_t)id * Dm))[t];
    v4bf o = {(__bf16)v.x, (__bf16)v.y, (__bf16)v.z, (__bf16)v.w};
    *(v4bf*)(Xlds + i * LDX + t * 4) = o;
  }
  __syncthreads();

  for (int dir = 0; dir < 2; ++dir) {
    const __bf16* Wgp = dir ? Wgp_b : Wgp_f;
    const float*  bg  = dir ? bg_b  : bg_f;
    const __bf16* WBp = dir ? WBp_b : WBp_f;
    float*        Bt  = dir ? Bt_b  : Bt_f;

    // ---- Stage A: Y[64,1024] = sigmoid(X Wg^T + bg) .* X (bf16 in LDS) ----
    {
      v8f acc[4][8] = {};
      for (int kt = 0; kt < Dm / 32; ++kt) {
        v16bf a0 = load_A(Xlds,            kt * 32, lane);
        v16bf a1 = load_A(Xlds + 16 * LDX, kt * 32, lane);
        v16bf a2 = load_A(Xlds + 32 * LDX, kt * 32, lane);
        v16bf a3 = load_A(Xlds + 48 * LDX, kt * 32, lane);
#pragma unroll
        for (int j = 0; j < 8; ++j) {
          v16bf bb = load_Bp(Wgp, wave + j * 8, kt, lane);
          acc[0][j] = __builtin_amdgcn_wmma_f32_16x16x32_bf16(
              false, a0, false, bb, (short)0, acc[0][j], false, false);
          acc[1][j] = __builtin_amdgcn_wmma_f32_16x16x32_bf16(
              false, a1, false, bb, (short)0, acc[1][j], false, false);
          acc[2][j] = __builtin_amdgcn_wmma_f32_16x16x32_bf16(
              false, a2, false, bb, (short)0, acc[2][j], false, false);
          acc[3][j] = __builtin_amdgcn_wmma_f32_16x16x32_bf16(
              false, a3, false, bb, (short)0, acc[3][j], false, false);
        }
      }
      const int mbase = (lane >> 4) * 8;
      const int n     = lane & 15;
#pragma unroll
      for (int j = 0; j < 8; ++j) {
        const int col = (wave + j * 8) * 16 + n;
        const float bgv = bg[col];
#pragma unroll
        for (int rt = 0; rt < 4; ++rt) {
#pragma unroll
          for (int r = 0; r < 8; ++r) {
            const int mr = rt * 16 + mbase + r;
            float xv  = (float)Xlds[mr * LDX + col];
            float pre = acc[rt][j][r] + bgv;
            float sig = 1.f / (1.f + __expf(-pre));
            Ylds[mr * LDX + col] = (__bf16)(sig * xv);
          }
        }
      }
    }
    __syncthreads();

    // ---- Stage B: Bt[64,16] = Y @ WB^T. Wave w: row-tile w/2, K half w&1.
    //      Cross-wave (pairwise) reduce in LDS. ----
    {
      const int rt = wave >> 1;
      const int kh = wave & 1;
      v8f acc2 = {};
#pragma unroll
      for (int kt = 0; kt < 16; ++kt) {
        const int kb = kh * 16 + kt;  // K fragment index 0..31
        v16bf a  = load_A(Ylds + rt * 16 * LDX, kb * 32, lane);
        v16bf bb = load_Bp(WBp, 0, kb, lane);
        acc2 = __builtin_amdgcn_wmma_f32_16x16x32_bf16(
            false, a, false, bb, (short)0, acc2, false, false);
      }
#pragma unroll
      for (int r = 0; r < 8; ++r) red[wave * 256 + r * 32 + lane] = acc2[r];
    }
    __syncthreads();
    {
      const int r  = t >> 5;
      const int ln = t & 31;
      const int m  = (ln >> 4) * 8 + r;
      const int n  = ln & 15;
#pragma unroll
      for (int rt = 0; rt < 4; ++rt) {
        float s = red[(2 * rt) * 256 + t] + red[(2 * rt + 1) * 256 + t];
        Bt[(m0 + rt * 16 + m) * Nn + n] = s;
      }
    }
    __syncthreads();  // LDS reused by next direction
  }
}

// ---------------------------------------------------------------------------
// Sequential recurrence: 1024 independent (dir,b,n) scans of 2048 tanh steps.
// Uses the CDNA5 hardware tanh transcendental to shorten the dependent chain.
__global__ void scan_kernel(const float* __restrict__ A_f, const float* __restrict__ A_b,
                            const float* __restrict__ Bt_f, const float* __restrict__ Bt_b,
                            float* __restrict__ h_comb) {
  const int t = blockIdx.x * blockDim.x + threadIdx.x;
  if (t >= 2 * Bb * Nn) return;
  const int dir = t >> 9;
  const int idx = t & 511;
  const int b = idx >> 4;
  const int n = idx & 15;
  const float* A  = dir ? A_b  : A_f;
  const float* Bt = dir ? Bt_b : Bt_f;

  float am = 0.f;
  for (int d = 0; d < Dm; ++d) am += A[d * Nn + n];
  am *= (1.f / (float)Dm);

  const float* base = Bt + ((size_t)b * Ss) * Nn + n;
  float h = 0.f;
  if (dir == 0) {
    for (int s = 0; s < Ss; ++s)      h = FAST_TANH(h * am + base[(size_t)s * Nn]);
  } else {
    for (int s = Ss - 1; s >= 0; --s) h = FAST_TANH(h * am + base[(size_t)s * Nn]);
  }
  h_comb[b * 32 + dir * 16 + n] = h;
}

// ---------------------------------------------------------------------------
// Head: W1 -> LayerNorm -> ReLU -> W2 -> ReLU -> Wh. One block per batch row.
__global__ void __launch_bounds__(256)
head_kernel(const float* __restrict__ h_comb,
            const float* __restrict__ W1, const float* __restrict__ b1,
            const float* __restrict__ ln_g, const float* __restrict__ ln_b,
            const float* __restrict__ W2, const float* __restrict__ b2,
            const float* __restrict__ Wh, const float* __restrict__ bh,
            float* __restrict__ out) {
  __shared__ float hrow[32];
  __shared__ float z[Dm];
  __shared__ float z2[Dm / 2];
  __shared__ float red[256];
  __shared__ float stat[2];
  const int t = threadIdx.x;
  const int b = blockIdx.x;

  if (t < 32) hrow[t] = h_comb[b * 32 + t];
  __syncthreads();

  for (int d = t; d < Dm; d += 256) {
    float acc = b1[d];
    const float* w = W1 + d * 32;
#pragma unroll 8
    for (int k = 0; k < 32; ++k) acc += hrow[k] * w[k];
    z[d] = acc;
  }
  __syncthreads();

  float s = 0.f;
  for (int d = t; d < Dm; d += 256) s += z[d];
  red[t] = s;
  __syncthreads();
  for (int off = 128; off > 0; off >>= 1) {
    if (t < off) red[t] += red[t + off];
    __syncthreads();
  }
  if (t == 0) stat[0] = red[0] * (1.f / (float)Dm);
  __syncthreads();
  const float mu = stat[0];

  s = 0.f;
  for (int d = t; d < Dm; d += 256) { float v = z[d] - mu; s += v * v; }
  red[t] = s;
  __syncthreads();
  for (int off = 128; off > 0; off >>= 1) {
    if (t < off) red[t] += red[t + off];
    __syncthreads();
  }
  if (t == 0) stat[1] = rsqrtf(red[0] * (1.f / (float)Dm) + 1e-5f);
  __syncthreads();
  const float rstd = stat[1];

  for (int d = t; d < Dm; d += 256) {
    float v = (z[d] - mu) * rstd * ln_g[d] + ln_b[d];
    z[d] = v > 0.f ? v : 0.f;
  }
  __syncthreads();

  for (int j = t; j < Dm / 2; j += 256) {
    float acc = b2[j];
    const float* w = W2 + (size_t)j * Dm;
    for (int d = 0; d < Dm; ++d) acc += z[d] * w[d];
    z2[j] = acc > 0.f ? acc : 0.f;
  }
  __syncthreads();

  if (t < 3) {
    float acc = bh[t];
    const float* w = Wh + t * (Dm / 2);
    for (int j = 0; j < Dm / 2; ++j) acc += z2[j] * w[j];
    out[b * 3 + t] = acc;
  }
}

// ---------------------------------------------------------------------------
extern "C" void kernel_launch(void* const* d_in, const int* in_sizes, int n_in,
                              void* d_out, int out_size, void* d_ws, size_t ws_size,
                              hipStream_t stream) {
  const int*   ids  = (const int*)d_in[0];
  const float* emb  = (const float*)d_in[1];
  const float* A_f  = (const float*)d_in[2];
  const float* Wg_f = (const float*)d_in[3];
  const float* bg_f = (const float*)d_in[4];
  const float* WB_f = (const float*)d_in[5];
  const float* A_b  = (const float*)d_in[6];
  const float* Wg_b = (const float*)d_in[7];
  const float* bg_b = (const float*)d_in[8];
  const float* WB_b = (const float*)d_in[9];
  const float* W1   = (const float*)d_in[10];
  const float* b1   = (const float*)d_in[11];
  const float* ln_g = (const float*)d_in[12];
  const float* ln_b = (const float*)d_in[13];
  const float* W2   = (const float*)d_in[14];
  const float* b2   = (const float*)d_in[15];
  const float* Wh   = (const float*)d_in[16];
  const float* bh   = (const float*)d_in[17];
  float* out = (float*)d_out;

  // Workspace layout (bytes):
  //   Wgp_f 2 MB | Wgp_b 2 MB | WBp_f 32 KB | WBp_b 32 KB |
  //   Bt_f 4 MB | Bt_b 4 MB | h_comb 4 KB        => ~12.1 MB total
  char* ws = (char*)d_ws;
  const size_t wg_sz = (size_t)Dm * Dm * sizeof(__bf16);   // 2 MB
  const size_t wb_sz = (size_t)Nn * Dm * sizeof(__bf16);   // 32 KB
  const size_t bt_sz = (size_t)BS * Nn * sizeof(float);    // 4 MB
  __bf16* Wgp_f = (__bf16*)(ws);
  __bf16* Wgp_b = (__bf16*)(ws + wg_sz);
  __bf16* WBp_f = (__bf16*)(ws + 2 * wg_sz);
  __bf16* WBp_b = (__bf16*)(ws + 2 * wg_sz + wb_sz);
  float*  Bt_f   = (float*)(ws + 2 * wg_sz + 2 * wb_sz);
  float*  Bt_b   = (float*)(ws + 2 * wg_sz + 2 * wb_sz + bt_sz);
  float*  h_comb = (float*)(ws + 2 * wg_sz + 2 * wb_sz + 2 * bt_sz);

  // One-time (per call) weight pack: f32 -> bf16 in WMMA B-fragment order.
  pack_b_frags<<<(Dm * Dm) / 256, 256, 0, stream>>>(Wg_f, Wgp_f);
  pack_b_frags<<<(Dm * Dm) / 256, 256, 0, stream>>>(Wg_b, Wgp_b);
  pack_b_frags<<<(Nn * Dm) / 256, 256, 0, stream>>>(WB_f, WBp_f);
  pack_b_frags<<<(Nn * Dm) / 256, 256, 0, stream>>>(WB_b, WBp_b);

  fused_gate_proj<<<BS / MT, 256, 0, stream>>>(ids, emb, Wgp_f, bg_f, WBp_f,
                                               Wgp_b, bg_b, WBp_b, Bt_f, Bt_b);
  scan_kernel<<<4, 256, 0, stream>>>(A_f, A_b, Bt_f, Bt_b, h_comb);
  head_kernel<<<Bb, 256, 0, stream>>>(h_comb, W1, b1, ln_g, ln_b, W2, b2, Wh, bh, out);
}
